// Net_71373766525077
// MI455X (gfx1250) — matrix operand order repaired and checked by reading, not compiled, not run on hardware
//
#include <hip/hip_runtime.h>

// SNN forward for MI455X (gfx1250):
//   Kernel 1: cur = x @ W^T, split-bf16 WMMA (x~hi+lo, W~hi+lo, 3 WMMAs/tile),
//             f32 tiles staged via Tensor Data Mover (TENSOR_LOAD_TO_LDS,
//             TENSORcnt-synced, double-buffered, overlapped with compute),
//             then converted in LDS to bf16 hi/lo planes.
//   Kernel 2: sequential LIF scan over T, in-place on mem_rec half of d_out.

#if defined(__has_builtin)
#if __has_builtin(__builtin_amdgcn_tensor_load_to_lds)
#define USE_TDM 1
#else
#define USE_TDM 0
#endif
#else
#define USE_TDM 0
#endif

#if defined(__has_include)
#if __has_include(<hip/amd_detail/amd_gfx1250_TDM.h>)
#define TDM_6ARG 1   // amdgpu-toolchain / therock-10.0 headers -> 6-arg builtin
#else
#define TDM_6ARG 0   // ROCm 7.2 -> 5-arg builtin
#endif
#else
#define TDM_6ARG 0
#endif

typedef __attribute__((ext_vector_type(16))) __bf16 v16bf;
typedef __attribute__((ext_vector_type(8)))  __bf16 v8bf;
typedef __attribute__((ext_vector_type(4)))  __bf16 v4bf;
typedef __attribute__((ext_vector_type(8)))  float  v8f;
typedef __attribute__((ext_vector_type(4)))  float  v4f;
typedef __attribute__((ext_vector_type(4)))  unsigned int v4u;
typedef __attribute__((ext_vector_type(8)))  int    v8i;
typedef __attribute__((ext_vector_type(4)))  int    v4i;

constexpr int T_DIM = 1024;   // timesteps (GEMM M)
constexpr int N_DIM = 4096;   // output neurons (GEMM N)
constexpr int K_DIM = 4096;   // input features (GEMM K)

constexpr int BM = 128;       // block tile M
constexpr int BN = 128;       // block tile N
constexpr int BK = 32;        // K step == WMMA bf16 K
constexpr int LDT = BK + 24;  // bf16 plane row stride: 56 bf16 = 112 B (7*16B)
constexpr int SFD = 36;       // raw f32 tile row stride in dwords (32 + 4 pad)
                              //  -> 144 B rows; rows 0..15 hit distinct banks

#if USE_TDM
// Issue one 2D TDM tile load: 32 f32 (contiguous) x 128 rows, row stride
// K_DIM elements, into LDS at lds_off with 4 pad dwords every 32 dwords
// (pad_interval code 4 = 32 dwords, pad_amount code 3 = 4 dwords -> SFD=36).
__device__ __forceinline__ void tdm_load_tile_f32(const float* gsrc,
                                                  unsigned lds_off,
                                                  unsigned tensor_rows) {
  unsigned long long ga = (unsigned long long)(const void*)gsrc;
  v4u g0;
  g0.x = 1u;                                   // count=1 (valid user D#)
  g0.y = lds_off;                              // lds_addr (bytes)
  g0.z = (unsigned)ga;                         // global_addr[31:0]
  g0.w = (unsigned)(ga >> 32) | 0x80000000u;   // global_addr[56:32] | type=2<<30
  v8i g1;
  g1[0] = (2 << 16)       // data_size = 4 bytes
        | (1 << 20)       // pad_enable
        | (4 << 22)       // pad_interval: code 4 = every 32 dwords
        | (3 << 25);      // pad_amount:   code 3 = 4 dwords
  g1[1] = (int)((K_DIM & 0xFFFF) << 16);                        // tensor_dim0 lo16
  g1[2] = (int)(((unsigned)K_DIM >> 16) |
                ((tensor_rows & 0xFFFFu) << 16));               // dim0 hi | dim1 lo
  g1[3] = (int)((tensor_rows >> 16) | ((unsigned)BK << 16));    // dim1 hi | tile_dim0
  g1[4] = (int)(BM & 0xFFFF);                                   // tile_dim1 | tile_dim2=0
  g1[5] = (int)K_DIM;                                           // tensor_dim0_stride lo32
  g1[6] = 0;                                                    // stride0 hi | stride1 lo
  g1[7] = 0;                                                    // stride1 hi
  v4i z4 = {0, 0, 0, 0};
#if TDM_6ARG
  v8i z8 = {0, 0, 0, 0, 0, 0, 0, 0};
  __builtin_amdgcn_tensor_load_to_lds(g0, g1, z4, z4, z8, 0);
#else
  __builtin_amdgcn_tensor_load_to_lds(g0, g1, z4, z4, 0);
#endif
}
#endif  // USE_TDM

__global__ __launch_bounds__(256)
void snn_gemm_bf16x3(const float* __restrict__ X,   // [T_DIM, K_DIM]
                     const float* __restrict__ W,   // [N_DIM, K_DIM]
                     float* __restrict__ Cur)       // [T_DIM, N_DIM]
{
  // Raw f32 tile buffers (TDM/async destination), double-buffered.
  __shared__ float Fa[2][BM * SFD];   // 2 x 18 KB
  __shared__ float Fb[2][BN * SFD];   // 2 x 18 KB
  // bf16 hi/lo planes (WMMA operands), single-buffered (barriers cover reuse).
  __shared__ __bf16 Ah[BM][LDT];
  __shared__ __bf16 Al[BM][LDT];
  __shared__ __bf16 Bh[BN][LDT];
  __shared__ __bf16 Bl[BN][LDT];

  const int tid  = threadIdx.x;
  const int lane = tid & 31;
  const int wave = tid >> 5;
  const int half = lane >> 4;   // K-half selector for A/B fragments
  const int l16  = lane & 15;   // row (A) / column (B,C) within 16

  // 8 waves -> 2 (M) x 4 (N); each wave computes a 64x32 output subtile.
  const int m0 = (wave >> 2) * 64;
  const int n0 = (wave & 3) * 32;

  const int gm0 = blockIdx.y * BM;
  const int gn0 = blockIdx.x * BN;

  // Stage/convert assignment: thread owns one 16-float row segment of A and B.
  const int lrow = tid >> 1;         // 0..127
  const int lseg = (tid & 1) * 16;   // 0 or 16

  v8f acc[4][2];
  #pragma unroll
  for (int i = 0; i < 4; ++i)
    #pragma unroll
    for (int j = 0; j < 2; ++j)
      acc[i][j] = {};

  constexpr int NT = K_DIM / BK;     // 128 K-steps

  auto issue_tiles = [&](int it) {
#if USE_TDM
    if (wave == 0) {   // EXEC ignored by TDM; one wave issues for the WG
      tdm_load_tile_f32(X + (size_t)gm0 * K_DIM + (size_t)it * BK,
                        (unsigned)(size_t)&Fa[it & 1][0], (unsigned)T_DIM);
      tdm_load_tile_f32(W + (size_t)gn0 * K_DIM + (size_t)it * BK,
                        (unsigned)(size_t)&Fb[it & 1][0], (unsigned)N_DIM);
    }
#else
    const float* ga = X + (size_t)(gm0 + lrow) * K_DIM + it * BK + lseg;
    const float* gb = W + (size_t)(gn0 + lrow) * K_DIM + it * BK + lseg;
    float* la = &Fa[it & 1][lrow * SFD + lseg];
    float* lb = &Fb[it & 1][lrow * SFD + lseg];
    #pragma unroll
    for (int u = 0; u < 4; ++u) {
      *(v4f*)(la + 4 * u) = *(const v4f*)(ga + 4 * u);
      *(v4f*)(lb + 4 * u) = *(const v4f*)(gb + 4 * u);
    }
#endif
  };

  // Prologue: tile 0.
  issue_tiles(0);
#if USE_TDM
  if (wave == 0) __builtin_amdgcn_s_wait_tensorcnt((short)0);
#endif
  __syncthreads();

  for (int it = 0; it < NT; ++it) {
    // Kick off DMA for tile it+1 (targets the other F buffer; its previous
    // readers finished before the end-of-iteration barrier of it-1).
    if (it + 1 < NT) issue_tiles(it + 1);

    // ---- convert stage: raw f32 tile -> bf16 hi/lo planes ----
    {
      const float* fa = &Fa[it & 1][lrow * SFD + lseg];
      const float* fb = &Fb[it & 1][lrow * SFD + lseg];
      #pragma unroll
      for (int u = 0; u < 4; ++u) {
        v4f xa = *(const v4f*)(fa + 4 * u);
        v4f xb = *(const v4f*)(fb + 4 * u);
        v4bf ah, al, bh, bl;
        #pragma unroll
        for (int e = 0; e < 4; ++e) {
          __bf16 h = (__bf16)xa[e];                 // RNE to bf16
          ah[e] = h;
          al[e] = (__bf16)(xa[e] - (float)h);       // residual (next 8 bits)
          __bf16 g = (__bf16)xb[e];
          bh[e] = g;
          bl[e] = (__bf16)(xb[e] - (float)g);
        }
        *(v4bf*)&Ah[lrow][lseg + 4 * u] = ah;
        *(v4bf*)&Al[lrow][lseg + 4 * u] = al;
        *(v4bf*)&Bh[lrow][lseg + 4 * u] = bh;
        *(v4bf*)&Bl[lrow][lseg + 4 * u] = bl;
      }
    }
    __syncthreads();   // planes visible; raw-tile reads done

    // ---- load WMMA fragments (ISA 16-bit A/B VGPR layouts) ----
    // A (16x32, MxK): lane m=l16; elems 0..7  = K[half*8 .. +7],
    //                              elems 8..15 = K[16+half*8 .. +7]
    // B (32x16, KxN): lane n=l16; elems 0..7  = K[half*8 .. +7],
    //                              elems 8..15 = K[16+half*8 .. +7]
    v16bf afh[4], afl[4], bfh[2], bfl[2];
    #pragma unroll
    for (int i = 0; i < 4; ++i) {
      const int row = m0 + i * 16 + l16;
      v8bf h0 = *(const v8bf*)&Ah[row][half * 8];
      v8bf h1 = *(const v8bf*)&Ah[row][16 + half * 8];
      afh[i] = __builtin_shufflevector(h0, h1, 0,1,2,3,4,5,6,7,8,9,10,11,12,13,14,15);
      v8bf l0 = *(const v8bf*)&Al[row][half * 8];
      v8bf l1 = *(const v8bf*)&Al[row][16 + half * 8];
      afl[i] = __builtin_shufflevector(l0, l1, 0,1,2,3,4,5,6,7,8,9,10,11,12,13,14,15);
    }
    #pragma unroll
    for (int j = 0; j < 2; ++j) {
      const int col = n0 + j * 16 + l16;
      v8bf h0 = *(const v8bf*)&Bh[col][half * 8];
      v8bf h1 = *(const v8bf*)&Bh[col][16 + half * 8];
      bfh[j] = __builtin_shufflevector(h0, h1, 0,1,2,3,4,5,6,7,8,9,10,11,12,13,14,15);
      v8bf l0 = *(const v8bf*)&Bl[col][half * 8];
      v8bf l1 = *(const v8bf*)&Bl[col][16 + half * 8];
      bfl[j] = __builtin_shufflevector(l0, l1, 0,1,2,3,4,5,6,7,8,9,10,11,12,13,14,15);
    }

    // ---- 8 independent C tiles x 3-term split-bf16 accumulation ----
    #pragma unroll
    for (int i = 0; i < 4; ++i)
      #pragma unroll
      for (int j = 0; j < 2; ++j) {
        acc[i][j] = __builtin_amdgcn_wmma_f32_16x16x32_bf16(
            false, afh[i], false, bfh[j], (short)0, acc[i][j], false, false);
        acc[i][j] = __builtin_amdgcn_wmma_f32_16x16x32_bf16(
            false, afh[i], false, bfl[j], (short)0, acc[i][j], false, false);
        acc[i][j] = __builtin_amdgcn_wmma_f32_16x16x32_bf16(
            false, afl[i], false, bfh[j], (short)0, acc[i][j], false, false);
      }

#if USE_TDM
    if (wave == 0 && it + 1 < NT) __builtin_amdgcn_s_wait_tensorcnt((short)0);
#endif
    __syncthreads();   // next raw tile visible; plane reads done
  }

  // ---- epilogue: C layout lane l -> (m = 8*(l>=16)+vgpr, n = l&15) ----
  #pragma unroll
  for (int i = 0; i < 4; ++i)
    #pragma unroll
    for (int j = 0; j < 2; ++j) {
      const int gm = gm0 + m0 + i * 16 + half * 8;
      const int gn = gn0 + n0 + j * 16 + l16;
      float* p = Cur + (size_t)gm * N_DIM + gn;
      #pragma unroll
      for (int r = 0; r < 8; ++r)
        p[(size_t)r * N_DIM] = acc[i][j][r];
    }
}

// Sequential LIF scan: one thread per neuron, T dependent steps.
// MemCur initially holds cur[t,o]; overwritten in place with mem_rec.
__global__ __launch_bounds__(64)
void snn_scan(float* __restrict__ MemCur,   // [T_DIM, N_DIM]
              float* __restrict__ Spk)      // [T_DIM, N_DIM]
{
  const int o = blockIdx.x * 64 + threadIdx.x;
  float mem = 0.0f;
  for (int t = 0; t < T_DIM; ++t) {
    const size_t idx = (size_t)t * N_DIM + o;
    const float c = MemCur[idx];
    const float reset = (mem > 1.0f) ? 1.0f : 0.0f;   // from PREVIOUS mem
    mem = 0.9f * mem + c - reset;                     // THRESHOLD == 1.0
    MemCur[idx] = mem;
    Spk[idx] = (mem > 1.0f) ? 1.0f : 0.0f;
  }
}

extern "C" void kernel_launch(void* const* d_in, const int* in_sizes, int n_in,
                              void* d_out, int out_size, void* d_ws, size_t ws_size,
                              hipStream_t stream) {
  (void)in_sizes; (void)n_in; (void)out_size; (void)d_ws; (void)ws_size;
  const float* x = (const float*)d_in[0];   // [1024, 4096]
  const float* w = (const float*)d_in[1];   // [4096, 4096]
  float* out = (float*)d_out;               // spk_rec ++ mem_rec
  float* spk = out;
  float* mem = out + (size_t)T_DIM * N_DIM;

  dim3 ggrid(N_DIM / BN, T_DIM / BM);       // 32 x 8 workgroups
  snn_gemm_bf16x3<<<ggrid, dim3(256), 0, stream>>>(x, w, mem);
  snn_scan<<<dim3(N_DIM / 64), dim3(64), 0, stream>>>(mem, spk);
}
